// Head_21414706937890
// MI455X (gfx1250) — matrix operand order repaired and checked by reading, not compiled, not run on hardware
//
#include <hip/hip_runtime.h>
#include <hip/hip_bf16.h>

typedef __attribute__((ext_vector_type(16))) _Float16 v16h;
typedef __attribute__((ext_vector_type(8)))  float    v8f;
typedef int vint4 __attribute__((vector_size(16)));   // b128 unit for async-LDS

#define T_SEQ 4096
#define EMB   1024
#define HS    64
#define BATCH 4

#if defined(__has_builtin)
#if __has_builtin(__builtin_amdgcn_global_load_async_to_lds_b128)
#define HAVE_ASYNC 1
#endif
#endif

// ---------- fragment helpers (ISA 16-bit WMMA layouts, wave32) ----------
// A (16x32, MxK): lane L holds row M=L&15; h=L>>4. Elements 0..7 = K kk+8h..,
// elements 8..15 = K kk+16+8h.. (per CDNA5 ISA A-matrix table).
__device__ __forceinline__ v16h a_frag_f32(const float* rowBase, int kk, int h) {
  v16h r;
  const float* p0 = rowBase + kk + 8 * h;
  const float* p1 = rowBase + kk + 16 + 8 * h;
#pragma unroll
  for (int j = 0; j < 8; ++j) { r[j] = (_Float16)p0[j]; r[8 + j] = (_Float16)p1[j]; }
  return r;
}
__device__ __forceinline__ v16h a_frag_f16(const _Float16* rowBase, int kk, int h) {
  v16h r;
  const _Float16* p0 = rowBase + kk + 8 * h;
  const _Float16* p1 = rowBase + kk + 16 + 8 * h;
#pragma unroll
  for (int j = 0; j < 8; ++j) { r[j] = p0[j]; r[8 + j] = p1[j]; }
  return r;
}
// B (32x16, KxN): lane L holds column N=L&15; h=L>>4. Elements 0..7 = K kk+8h..,
// elements 8..15 = K kk+16+8h.. (same K order as the A fragment).
__device__ __forceinline__ v16h b_frag_f16(const _Float16* colBase, int kk, int h) {
  v16h r;
  const _Float16* p0 = colBase + kk + 8 * h;
  const _Float16* p1 = colBase + kk + 16 + 8 * h;
#pragma unroll
  for (int j = 0; j < 8; ++j) { r[j] = p0[j]; r[8 + j] = p1[j]; }
  return r;
}
__device__ __forceinline__ v8f zero8() {
  v8f z;
#pragma unroll
  for (int i = 0; i < 8; ++i) z[i] = 0.0f;
  return z;
}
__device__ __forceinline__ v8f wmma16(v16h a, v16h b, v8f c) {
  return __builtin_amdgcn_wmma_f32_16x16x32_f16(false, a, false, b, (short)0, c,
                                                false, false);
}

// 16B global -> LDS staging chunk (async-to-LDS if the toolchain exposes it)
__device__ __forceinline__ void stage16(const _Float16* g, _Float16* l) {
#ifdef HAVE_ASYNC
  __builtin_amdgcn_global_load_async_to_lds_b128(
      (__attribute__((address_space(1))) vint4*)(g),
      (__attribute__((address_space(3))) vint4*)(l), 0, 0);
#else
  *reinterpret_cast<float4*>(l) = *reinterpret_cast<const float4*>(g);
#endif
}
__device__ __forceinline__ void stage_wait() {
#ifdef HAVE_ASYNC
#if __has_builtin(__builtin_amdgcn_s_wait_asynccnt)
  __builtin_amdgcn_s_wait_asynccnt(0);
#else
  asm volatile("s_wait_asynccnt 0" ::: "memory");
#endif
#endif
}

// ------------- kernel 0: one-time weight conversion to f16 --------------
// w16 layout: [3][HS][EMB] f16 (q, k, v weight rows).
__global__ __launch_bounds__(256) void wconv_kernel(
    const float* __restrict__ Wq, const float* __restrict__ Wk,
    const float* __restrict__ Wv, _Float16* __restrict__ w16) {
  const int i = blockIdx.x * 256 + threadIdx.x;  // 3*HS*EMB = 196608 total
  const float* src = (i < HS * EMB) ? Wq : (i < 2 * HS * EMB) ? Wk : Wv;
  w16[i] = (_Float16)src[i & (HS * EMB - 1)];
}

// ---------------- kernel 1: fused QKV projection + RoPE -----------------
// grid = rows/128; block = 256 (8 wave32). Wave w -> one 16-row tile,
// computing q, k and v for it (x loaded/converted exactly once).
__global__ __launch_bounds__(256) void qkv_rope_kernel(
    const float* __restrict__ x, const _Float16* __restrict__ w16,
    _Float16* __restrict__ qf, _Float16* __restrict__ kf,
    _Float16* __restrict__ vT) {
  const int lane = threadIdx.x & 31;
  const int wave = threadIdx.x >> 5;
  const int h  = lane >> 4;
  const int nl = lane & 15;
  const int rowBase = (blockIdx.x * 8 + wave) * 16;  // global row in [0, B*T)

  const float* xRow = x + (size_t)(rowBase + nl) * EMB;

  v8f acc[3][4];
#pragma unroll
  for (int m = 0; m < 3; ++m)
#pragma unroll
    for (int nt = 0; nt < 4; ++nt) acc[m][nt] = zero8();

  for (int kk = 0; kk < EMB; kk += 32) {
    v16h a = a_frag_f32(xRow, kk, h);
#pragma unroll
    for (int m = 0; m < 3; ++m) {
#pragma unroll
      for (int nt = 0; nt < 4; ++nt) {
        const _Float16* wCol = w16 + (size_t)(m * HS + nt * 16 + nl) * EMB;
        acc[m][nt] = wmma16(a, b_frag_f16(wCol, kk, h), acc[m][nt]);
      }
    }
  }

  const int tBase = rowBase & (T_SEQ - 1);  // position within sequence

  // RoPE on q and k; in C layout the (even,odd) head-dim pair sits in
  // adjacent lanes -> one shfl_xor(1) exchange.
#pragma unroll
  for (int m = 0; m < 2; ++m) {
    _Float16* dst = (m == 0) ? qf : kf;
#pragma unroll
    for (int nt = 0; nt < 4; ++nt) {
      const int col  = nt * 16 + nl;
      const int pi   = col >> 1;
      const float freq = __powf(10000.0f, -(float)pi * (1.0f / 32.0f));
      const bool even = (col & 1) == 0;
#pragma unroll
      for (int r = 0; r < 8; ++r) {
        const int t = tBase + r + 8 * h;
        float sn, cs;
        __sincosf((float)t * freq, &sn, &cs);
        float v0 = acc[m][nt][r];
        float vp = __shfl_xor(v0, 1, 32);
        float rv = even ? (v0 * cs - vp * sn) : (vp * sn + v0 * cs);
        dst[(size_t)(rowBase + r + 8 * h) * HS + col] = (_Float16)rv;
      }
    }
  }

  // V stored transposed: vT[b][hs][t] so attention B-fragments are contiguous.
  const int bIdx = rowBase / T_SEQ;
#pragma unroll
  for (int nt = 0; nt < 4; ++nt) {
    _Float16* dstCol = vT + ((size_t)bIdx * HS + nt * 16 + nl) * T_SEQ;
#pragma unroll
    for (int r = 0; r < 8; ++r) dstCol[tBase + r + 8 * h] = (_Float16)acc[2][nt][r];
  }
}

// ---------------- kernel 2: causal flash attention ----------------------
// WG = 128 consecutive query rows (8 waves x 16). All waves share K/V kv
// blocks of 32 staged in LDS (async-to-LDS when available); each wave runs
// online softmax on its own 16-row tile, P transposed via per-wave LDS.
__global__ __launch_bounds__(256) void flash_attn_kernel(
    const _Float16* __restrict__ qf, const _Float16* __restrict__ kf,
    const _Float16* __restrict__ vT, float* __restrict__ out) {
  __shared__ __align__(16) _Float16 kLds[32 * 64];       // kv row x hs (4 KB)
  __shared__ __align__(16) _Float16 vLds[64 * 32];       // hs x kv row (4 KB)
  __shared__ __align__(16) _Float16 pbuf[8][16][32];     // per-wave P (8 KB)
  const int tid  = threadIdx.x;
  const int lane = tid & 31;
  const int wave = tid >> 5;
  const int h  = lane >> 4;
  const int nl = lane & 15;
  const int b       = blockIdx.x >> 5;          // 32 WGs per batch
  const int qBaseWg = (blockIdx.x & 31) * 128;  // WG's first query row
  const int qBase   = qBaseWg + wave * 16;      // this wave's query tile

  const _Float16* qRow = qf + (size_t)(b * T_SEQ + qBase + nl) * HS;
  const v16h qA0 = a_frag_f16(qRow, 0, h);
  const v16h qA1 = a_frag_f16(qRow, 32, h);

  v8f accO[4];
#pragma unroll
  for (int nt = 0; nt < 4; ++nt) accO[nt] = zero8();
  float mi[8], li[8];
#pragma unroll
  for (int r = 0; r < 8; ++r) { mi[r] = -1e30f; li[r] = 0.0f; }

  const float scale = 0.125f;  // HS^-0.5
  // staging assignment (one 16B chunk of K and one of V per thread)
  const int krow = tid >> 3, kch = tid & 7;   // 32 rows x 8 chunks
  const int vrow = tid >> 2, vch = tid & 3;   // 64 rows x 4 chunks

  const int cMax = qBaseWg + 127;
  for (int c = 0; c <= cMax; c += 32) {
    stage16(kf + (size_t)(b * T_SEQ + c + krow) * HS + kch * 8,
            &kLds[krow * 64 + kch * 8]);
    stage16(vT + (size_t)(b * HS + vrow) * T_SEQ + c + vch * 8,
            &vLds[vrow * 32 + vch * 8]);
    stage_wait();
    __syncthreads();

    if (c <= qBase + 15) {
      // S = Q K^T for two 16-wide kv tiles (K fragments from LDS)
      v8f s0 = zero8(), s1 = zero8();
      s0 = wmma16(qA0, b_frag_f16(&kLds[nl * 64], 0, h),  s0);
      s0 = wmma16(qA1, b_frag_f16(&kLds[nl * 64], 32, h), s0);
      s1 = wmma16(qA0, b_frag_f16(&kLds[(16 + nl) * 64], 0, h),  s1);
      s1 = wmma16(qA1, b_frag_f16(&kLds[(16 + nl) * 64], 32, h), s1);
#pragma unroll
      for (int r = 0; r < 8; ++r) { s0[r] *= scale; s1[r] *= scale; }

      if (c + 31 > qBase) {  // block touches the diagonal -> causal mask
#pragma unroll
        for (int r = 0; r < 8; ++r) {
          const int q = qBase + r + 8 * h;
          if (c + nl      > q) s0[r] = -1e30f;
          if (c + 16 + nl > q) s1[r] = -1e30f;
        }
      }

      // online softmax; each (r, half) is one row, reduce over 16 lanes
#pragma unroll
      for (int r = 0; r < 8; ++r) {
        float v = fmaxf(s0[r], s1[r]);
        v = fmaxf(v, __shfl_xor(v, 1, 32));
        v = fmaxf(v, __shfl_xor(v, 2, 32));
        v = fmaxf(v, __shfl_xor(v, 4, 32));
        v = fmaxf(v, __shfl_xor(v, 8, 32));
        const float mNew = fmaxf(mi[r], v);
        const float corr = __expf(mi[r] - mNew);
        const float p0 = __expf(s0[r] - mNew);
        const float p1 = __expf(s1[r] - mNew);
        float ps = p0 + p1;
        ps += __shfl_xor(ps, 1, 32);
        ps += __shfl_xor(ps, 2, 32);
        ps += __shfl_xor(ps, 4, 32);
        ps += __shfl_xor(ps, 8, 32);
        li[r] = li[r] * corr + ps;
        mi[r] = mNew;
#pragma unroll
        for (int nt = 0; nt < 4; ++nt) accO[nt][r] *= corr;
        const int mrow = r + 8 * h;
        pbuf[wave][mrow][nl]      = (_Float16)p0;  // C layout -> LDS
        pbuf[wave][mrow][16 + nl] = (_Float16)p1;
      }

      // reload P as A fragment (same-wave LDS is in-order) and do P*V
      const v16h pA = a_frag_f16(&pbuf[wave][nl][0], 0, h);
#pragma unroll
      for (int nt = 0; nt < 4; ++nt)
        accO[nt] = wmma16(pA, b_frag_f16(&vLds[(nt * 16 + nl) * 32], 0, h),
                          accO[nt]);
    }
    __syncthreads();  // protect kLds/vLds before next stage
  }

  // epilogue: out = accO / l
#pragma unroll
  for (int nt = 0; nt < 4; ++nt) {
#pragma unroll
    for (int r = 0; r < 8; ++r) {
      out[(size_t)(b * T_SEQ + qBase + r + 8 * h) * HS + nt * 16 + nl] =
          accO[nt][r] / li[r];
    }
  }
}

extern "C" void kernel_launch(void* const* d_in, const int* in_sizes, int n_in,
                              void* d_out, int out_size, void* d_ws,
                              size_t ws_size, hipStream_t stream) {
  (void)in_sizes; (void)n_in; (void)out_size; (void)ws_size;
  const float* x  = (const float*)d_in[0];
  const float* Wq = (const float*)d_in[1];
  const float* Wk = (const float*)d_in[2];
  const float* Wv = (const float*)d_in[3];
  float* out = (float*)d_out;

  _Float16* qf  = (_Float16*)d_ws;                       // 2 MB
  _Float16* kf  = qf + (size_t)BATCH * T_SEQ * HS;       // 2 MB
  _Float16* vT  = kf + (size_t)BATCH * T_SEQ * HS;       // 2 MB (transposed)
  _Float16* w16 = vT + (size_t)BATCH * T_SEQ * HS;       // 384 KB

  wconv_kernel<<<(3 * HS * EMB) / 256, 256, 0, stream>>>(Wq, Wk, Wv, w16);

  dim3 g1((BATCH * T_SEQ) / (16 * 8));  // 128
  qkv_rope_kernel<<<g1, 256, 0, stream>>>(x, w16, qf, kf, vT);

  dim3 g2((BATCH * T_SEQ) / (16 * 8));  // 128 (32 WGs per batch)
  flash_attn_kernel<<<g2, 256, 0, stream>>>(qf, kf, vT, out);
}
